// TemporalAttention_10599979287213
// MI455X (gfx1250) — matrix operand order repaired and checked by reading, not compiled, not run on hardware
//
#include <hip/hip_runtime.h>

typedef _Float16 f16;
typedef __attribute__((ext_vector_type(16))) _Float16 v16h;
typedef __attribute__((ext_vector_type(8)))  _Float16 v8h;
typedef __attribute__((ext_vector_type(8)))  float    v8f;
typedef __attribute__((ext_vector_type(4)))  int      v4i;

#define B_ 16
#define C_ 512
#define T_ 1024
#define D_ 64

#ifndef __has_builtin
#define __has_builtin(x) 0
#endif

#if __has_builtin(__builtin_amdgcn_global_load_async_to_lds_b128)
#define USE_ASYNC 1
#else
#define USE_ASYNC 0
#endif

#if __has_builtin(__builtin_amdgcn_s_wait_asynccnt)
#define WAIT_ASYNC(n) __builtin_amdgcn_s_wait_asynccnt(n)
#else
#define WAIT_ASYNC(n) asm volatile("s_wait_asynccnt " #n ::: "memory")
#endif

// Copy 16 bytes global -> LDS. Async (ASYNCcnt-tracked, no VGPR data) when the
// gfx1250 builtin is available; otherwise synchronous through VGPRs.
// Builtin signature (from hipcc diagnostic): (v4i as1*, v4i as3*, imm, cpol).
__device__ __forceinline__ void copy16_g2l(const f16* g, f16* l) {
#if USE_ASYNC
  __builtin_amdgcn_global_load_async_to_lds_b128(
      (__attribute__((address_space(1))) v4i*)(g),
      (__attribute__((address_space(3))) v4i*)(l), 0, 0);
#else
  *(v8h*)l = *(const v8h*)g;
#endif
}

__device__ inline v16h cvt16(const float4 a, const float4 b,
                             const float4 c, const float4 d) {
  v16h r;
  r[0]=(f16)a.x;  r[1]=(f16)a.y;  r[2]=(f16)a.z;  r[3]=(f16)a.w;
  r[4]=(f16)b.x;  r[5]=(f16)b.y;  r[6]=(f16)b.z;  r[7]=(f16)b.w;
  r[8]=(f16)c.x;  r[9]=(f16)c.y;  r[10]=(f16)c.z; r[11]=(f16)c.w;
  r[12]=(f16)d.x; r[13]=(f16)d.y; r[14]=(f16)d.z; r[15]=(f16)d.w;
  return r;
}

// ---------------------------------------------------------------------------
// Kernel 1: fused QKV projection.  Out[r,t] = sum_c W[r,c]*x[b,c,t] + bias[r]
// ---------------------------------------------------------------------------
__global__ __launch_bounds__(256)
void qkv_kernel(const float* __restrict__ x,
                const float* __restrict__ qw, const float* __restrict__ qb,
                const float* __restrict__ kw, const float* __restrict__ kb,
                const float* __restrict__ vw, const float* __restrict__ vb,
                f16* __restrict__ q_ws, f16* __restrict__ k_ws,
                f16* __restrict__ v_ws)
{
  const int lane = threadIdx.x & 31;
  const int wid  = blockIdx.x * 8 + (threadIdx.x >> 5);
  const int b    = wid / (40 * 64);
  const int mt   = (wid / 64) % 40;
  const int nt   = wid % 64;
  const int hi   = lane >> 4;
  const int lm   = lane & 15;

  const float* W; const float* bias; int roff;
  if (mt < 4)      { W = qw; bias = qb; roff = mt * 16; }
  else if (mt < 8) { W = kw; bias = kb; roff = (mt - 4) * 16; }
  else             { W = vw; bias = vb; roff = (mt - 8) * 16; }

  const float* wrow  = W + (size_t)(roff + lm) * C_;
  const float* xbase = x + (size_t)b * C_ * T_ + nt * 16;

  v8f acc = {};
  for (int kk = 0; kk < C_ / 32; ++kk) {
    const int kbase = kk * 32;
    const float4 wa0 = *(const float4*)(wrow + kbase + 8 * hi);
    const float4 wa1 = *(const float4*)(wrow + kbase + 8 * hi + 4);
    const float4 wa2 = *(const float4*)(wrow + kbase + 8 * hi + 16);
    const float4 wa3 = *(const float4*)(wrow + kbase + 8 * hi + 20);
    const v16h a = cvt16(wa0, wa1, wa2, wa3);
    const float* xr = xbase + lm + (size_t)(kbase + 8 * hi) * T_;
    __builtin_prefetch(xr + 32 * T_, 0, 0);
    const float4 xb0 = make_float4(xr[0], xr[T_], xr[2 * T_], xr[3 * T_]);
    const float4 xb1 = make_float4(xr[4 * T_], xr[5 * T_], xr[6 * T_], xr[7 * T_]);
    const float4 xb2 = make_float4(xr[16 * T_], xr[17 * T_], xr[18 * T_], xr[19 * T_]);
    const float4 xb3 = make_float4(xr[20 * T_], xr[21 * T_], xr[22 * T_], xr[23 * T_]);
    const v16h bf = cvt16(xb0, xb1, xb2, xb3);
    acc = __builtin_amdgcn_wmma_f32_16x16x32_f16(
        false, a, false, bf, (short)0, acc, false, false);
  }

  const int t = nt * 16 + lm;
  if (mt < 4) {
    f16* dst = q_ws + ((size_t)b * T_ + t) * D_;
    for (int v = 0; v < 8; ++v) {
      const int d = roff + v + 8 * hi;
      dst[d] = (f16)(acc[v] + bias[d]);
    }
  } else if (mt < 8) {
    for (int v = 0; v < 8; ++v) {
      const int d = roff + v + 8 * hi;
      k_ws[((size_t)b * D_ + d) * T_ + t] = (f16)(acc[v] + bias[d]);
    }
  } else {
    f16* dst = v_ws + ((size_t)b * T_ + t) * C_;
    for (int v = 0; v < 8; ++v) {
      const int c = roff + v + 8 * hi;
      dst[c] = (f16)(acc[v] + bias[c]);
    }
  }
}

// ---------------------------------------------------------------------------
// Kernel 2: energy = Q^T K + softmax stats; stores exp(S-rowmax) f16 + rowsum
// ---------------------------------------------------------------------------
__global__ __launch_bounds__(256)
void attn_softmax_kernel(const f16* __restrict__ q_ws,
                         const f16* __restrict__ k_ws,
                         f16* __restrict__ p_ws, float* __restrict__ l_ws)
{
  __shared__ float redmax[8][16];
  __shared__ float redsum[8][16];

  const int lane = threadIdx.x & 31;
  const int w    = threadIdx.x >> 5;
  const int b    = blockIdx.x >> 6;
  const int t0   = (blockIdx.x & 63) * 16;
  const int hi   = lane >> 4;
  const int lm   = lane & 15;

  const f16* qrow = q_ws + ((size_t)b * T_ + t0 + lm) * D_;
  v16h a0, a1;
  {
    const v8h l0 = *(const v8h*)(qrow + 8 * hi);
    const v8h h0 = *(const v8h*)(qrow + 8 * hi + 16);
    const v8h l1 = *(const v8h*)(qrow + 32 + 8 * hi);
    const v8h h1 = *(const v8h*)(qrow + 32 + 8 * hi + 16);
    for (int e = 0; e < 8; ++e) {
      a0[e] = l0[e]; a0[e + 8] = h0[e];
      a1[e] = l1[e]; a1[e + 8] = h1[e];
    }
  }

  const f16* kb0 = k_ws + ((size_t)b * D_ + 8 * hi) * T_ + lm;
  const f16* kb1 = k_ws + ((size_t)b * D_ + 32 + 8 * hi) * T_ + lm;

  v8f s[8];
  for (int st = 0; st < 8; ++st) {
    const int s0 = w * 128 + st * 16;
    v16h bf0, bf1;
    for (int e = 0; e < 16; ++e) {
      const int kr = (e & 7) + ((e >> 3) << 4);
      bf0[e] = kb0[(size_t)kr * T_ + s0];
      bf1[e] = kb1[(size_t)kr * T_ + s0];
    }
    v8f acc = {};
    acc = __builtin_amdgcn_wmma_f32_16x16x32_f16(
        false, a0, false, bf0, (short)0, acc, false, false);
    acc = __builtin_amdgcn_wmma_f32_16x16x32_f16(
        false, a1, false, bf1, (short)0, acc, false, false);
    s[st] = acc;
  }

  float gmax[8];
  for (int v = 0; v < 8; ++v) {
    float m = s[0][v];
    for (int st = 1; st < 8; ++st) m = fmaxf(m, s[st][v]);
    for (int o = 1; o < 16; o <<= 1) m = fmaxf(m, __shfl_xor(m, o, 32));
    gmax[v] = m;
  }
  if (lm == 0)
    for (int v = 0; v < 8; ++v) redmax[w][v + 8 * hi] = gmax[v];
  __syncthreads();
  for (int v = 0; v < 8; ++v) {
    float m = redmax[0][v + 8 * hi];
    for (int ww = 1; ww < 8; ++ww) m = fmaxf(m, redmax[ww][v + 8 * hi]);
    gmax[v] = m;
  }

  f16* pbase = p_ws + ((size_t)b * T_ + t0) * T_;
  float lsum[8];
  for (int v = 0; v < 8; ++v) lsum[v] = 0.f;
  for (int st = 0; st < 8; ++st) {
    const int scol = w * 128 + st * 16 + lm;
    for (int v = 0; v < 8; ++v) {
      const float e = __expf(s[st][v] - gmax[v]);
      lsum[v] += e;
      pbase[(size_t)(v + 8 * hi) * T_ + scol] = (f16)e;
    }
  }
  for (int v = 0; v < 8; ++v) {
    float ss = lsum[v];
    for (int o = 1; o < 16; o <<= 1) ss += __shfl_xor(ss, o, 32);
    if (lm == 0) redsum[w][v + 8 * hi] = ss;
  }
  __syncthreads();
  if (threadIdx.x < 16) {
    float l = 0.f;
    for (int ww = 0; ww < 8; ++ww) l += redsum[ww][threadIdx.x];
    l_ws[(size_t)b * T_ + t0 + threadIdx.x] = l;
  }
}

// ---------------------------------------------------------------------------
// Kernel 3: O[t,c] = sum_s P[t,s]*V[s,c];  out[b,c,t] = gamma*O/l + x[b,c,t]
// Workgroup (8 waves) computes 128t x 64c; K=1024 in 32-steps.
// P/V tiles staged in LDS via async global->LDS copies, double-buffered.
// Output bounced through LDS for coalesced float4 stores.
// ---------------------------------------------------------------------------
#define PT_STRIDE 40   // halves per P row (32 used + pad), 80B, 16B-aligned
#define VT_STRIDE 80   // halves per V row (64 used + pad), 160B, 32B-aligned

__global__ __launch_bounds__(256)
void pv_out_kernel(const f16* __restrict__ p_ws, const f16* __restrict__ v_ws,
                   const float* __restrict__ l_ws, const float* __restrict__ x,
                   const float* __restrict__ gamma_p, float* __restrict__ out)
{
  __shared__ f16 Pt[2][128][PT_STRIDE];   // 20480 B
  __shared__ f16 Vt[2][32][VT_STRIDE];    // 10240 B
  __shared__ float Ot[64][129];           // 33024 B (stride 129: conflict-free)

  const int tid  = threadIdx.x;
  const int lane = tid & 31;
  const int w    = tid >> 5;
  const int hi   = lane >> 4;
  const int lm   = lane & 15;

  const int b  = blockIdx.x >> 6;
  const int tb = (blockIdx.x >> 3) & 7;   // t block (128 rows)
  const int cb = blockIdx.x & 7;          // c block (64 cols)
  const size_t bT    = (size_t)b * T_;
  const int    tb128 = tb * 128;
  const int    cb64  = cb * 64;

  // per-thread async copy descriptors (3 x 16B per k-step)
  const int pr0 = tid >> 2,        ps0 = (tid & 3) * 8;        // P rows 0..63
  const int pr1 = 64 + (tid >> 2);                             // P rows 64..127
  const int vr  = tid >> 3,        vs  = (tid & 7) * 8;        // V rows 0..31

  const f16* pg0 = p_ws + (bT + tb128 + pr0) * T_ + ps0;
  const f16* pg1 = p_ws + (bT + tb128 + pr1) * T_ + ps0;
  const f16* vg  = v_ws + (bT + vr) * C_ + cb64 + vs;

#define ISSUE(k)                                                  \
  {                                                               \
    const int kb_ = (k) * 32;                                     \
    const int bf_ = (k) & 1;                                      \
    copy16_g2l(pg0 + kb_, &Pt[bf_][pr0][ps0]);                    \
    copy16_g2l(pg1 + kb_, &Pt[bf_][pr1][ps0]);                    \
    copy16_g2l(vg + (size_t)kb_ * C_, &Vt[bf_][vr][vs]);          \
  }

  v8f acc[4] = {};
  ISSUE(0);
  for (int k = 0; k < T_ / 32; ++k) {
    if (k + 1 < T_ / 32) {
      ISSUE(k + 1);
      WAIT_ASYNC(3);
    } else {
      WAIT_ASYNC(0);
    }
    __syncthreads();

    const int buf = k & 1;
    // A fragment: this wave's 16 t-rows, K = 32
    const f16* arow = &Pt[buf][w * 16 + lm][0];
    v16h a;
    {
      const v8h lo = *(const __attribute__((address_space(3))) v8h*)
                         (__attribute__((address_space(3))) const f16*)(arow + 8 * hi);
      const v8h h8 = *(const __attribute__((address_space(3))) v8h*)
                         (__attribute__((address_space(3))) const f16*)(arow + 8 * hi + 16);
      for (int e = 0; e < 8; ++e) { a[e] = lo[e]; a[e + 8] = h8[e]; }
    }
    // B fragments: lane holds c-col lm, s-rows in table order; 4 c-tiles
    const f16* brow = &Vt[buf][8 * hi][lm];
    for (int ct = 0; ct < 4; ++ct) {
      v16h bf;
      for (int e = 0; e < 16; ++e)
        bf[e] = brow[((e & 7) + ((e >> 3) << 4)) * VT_STRIDE + ct * 16];
      acc[ct] = __builtin_amdgcn_wmma_f32_16x16x32_f16(
          false, a, false, bf, (short)0, acc[ct], false, false);
    }
    __syncthreads();
  }

  // scale by gamma / l and stage output tile in LDS (c-major rows)
  const float g = *gamma_p;
  float invl[8];
  for (int v = 0; v < 8; ++v)
    invl[v] = g / l_ws[bT + tb128 + w * 16 + v + 8 * hi];
  for (int ct = 0; ct < 4; ++ct)
    for (int v = 0; v < 8; ++v)
      Ot[ct * 16 + lm][w * 16 + v + 8 * hi] = acc[ct][v] * invl[v];
  __syncthreads();

  // coalesced final store: thread -> (c = tid/4, 32-wide t chunk)
  const int c   = tid >> 2;
  const int t0l = (tid & 3) * 32;
  const size_t base = ((size_t)(b * C_ + cb64 + c)) * T_ + tb128 + t0l;
  for (int j = 0; j < 32; j += 4) {
    const float4 xv = *(const float4*)(x + base + j);
    float4 o;
    o.x = Ot[c][t0l + j + 0] + xv.x;
    o.y = Ot[c][t0l + j + 1] + xv.y;
    o.z = Ot[c][t0l + j + 2] + xv.z;
    o.w = Ot[c][t0l + j + 3] + xv.w;
    *(float4*)(out + base + j) = o;
  }
#undef ISSUE
}

// ---------------------------------------------------------------------------
extern "C" void kernel_launch(void* const* d_in, const int* in_sizes, int n_in,
                              void* d_out, int out_size, void* d_ws,
                              size_t ws_size, hipStream_t stream) {
  (void)in_sizes; (void)n_in; (void)out_size; (void)ws_size;
  const float* x     = (const float*)d_in[0];
  const float* qw    = (const float*)d_in[1];
  const float* qb    = (const float*)d_in[2];
  const float* kw    = (const float*)d_in[3];
  const float* kb    = (const float*)d_in[4];
  const float* vw    = (const float*)d_in[5];
  const float* vb    = (const float*)d_in[6];
  const float* gamma = (const float*)d_in[7];
  float* out = (float*)d_out;

  char* ws = (char*)d_ws;
  f16*   q_ws = (f16*)(ws);                          //  2 MB: [B,T,64]
  f16*   k_ws = (f16*)(ws + (size_t)(2)  * 1048576); //  2 MB: [B,64,T]
  f16*   v_ws = (f16*)(ws + (size_t)(4)  * 1048576); // 16 MB: [B,T,C]
  f16*   p_ws = (f16*)(ws + (size_t)(20) * 1048576); // 32 MB: [B,T,T]
  float* l_ws = (float*)(ws + (size_t)(52) * 1048576); // 64 KB: [B,T]

  qkv_kernel<<<5120, 256, 0, stream>>>(x, qw, qb, kw, kb, vw, vb,
                                       q_ws, k_ws, v_ws);
  attn_softmax_kernel<<<1024, 256, 0, stream>>>(q_ws, k_ws, p_ws, l_ws);
  // B(16) x T/128(8) x C/64(8) workgroups
  pv_out_kernel<<<1024, 256, 0, stream>>>(p_ws, v_ws, l_ws, x, gamma, out);
}
